// AFNO2D_2250562863951
// MI455X (gfx1250) — matrix operand order repaired and checked by reading, not compiled, not run on hardware
//
#include <hip/hip_runtime.h>

// ---------------------------------------------------------------------------
// AFNO2D on MI455X (gfx1250): DHT-as-GEMM + block-diagonal MLP, all on
// v_wmma_f32_16x16x32_bf16 (bf16 operands, f32 accumulate), wave32.
// ---------------------------------------------------------------------------

typedef __bf16 bf16_t;
typedef bf16_t v16bf __attribute__((ext_vector_type(16)));
typedef bf16_t v2bf  __attribute__((ext_vector_type(2)));
typedef float  v8f   __attribute__((ext_vector_type(8)));

struct alignas(16) U4 { unsigned int x, y, z, w; };
struct alignas(8)  U2 { unsigned int x, y; };
struct alignas(16) F4 { float v[4]; };

union Frag16 { v16bf v; U4 q[2]; };

// packed f32 -> bf16 pair; native __bf16 casts lower to v_cvt_pk_bf16_f32
__device__ __forceinline__ unsigned int pack2bf(float a, float b) {
  union { v2bf v; unsigned int u; } c;
  c.v[0] = (bf16_t)a;
  c.v[1] = (bf16_t)b;
  return c.u;
}

#define TB 256
#define BM 128
#define BN 128
#define BK 32
#define LP 56                             // LDS row stride in shorts (112B)

// Problem dims
#define DB 2
#define DH 128
#define DW 256
#define DC 768
#define KEPT 65
#define MR (DB * DH * KEPT)              // 16640 kept rows
static const long NTOT = (long)DB * DH * DW * DC;   // 50331648

// ---------------------------------------------------------------------------
// Generic batched strided GEMM:  D = alpha*(A@B) + beta*D + bias, opt. ReLU
// A: M x K (row stride lda), B: K x N (row stride ldb), D: M x N (ldd)
// Requires: K % 4 == 0, strides multiple of 4 floats (true for all call sites)
// Block tile 128x128, 8 waves, wave tile 32x64 -> 8 WMMAs per K-stage per wave
// ---------------------------------------------------------------------------
__global__ __launch_bounds__(TB) void k_gemm(
    const float* __restrict__ A, const float* __restrict__ Bm,
    float* __restrict__ D,
    int M, int N, int K,
    long lda, long ldb, long ldd,
    long bA, long bB, long bD,
    const float* __restrict__ bias, long biasB,
    float alpha, float beta, int act)
{
  __shared__ alignas(16) unsigned short As[BM][LP];    // A tile, [m][k]
  __shared__ alignas(16) unsigned short BsT[BN][LP];   // B tile transposed, [n][k]

  const int tid  = threadIdx.x;
  const int lane = tid & 31;
  const int wv   = tid >> 5;          // 8 waves / block
  const int half = lane >> 4;         // lane 0-15 vs 16-31
  const int lm   = lane & 15;
  const int wvM  = wv & 3;            // 4 wave-rows of 32
  const int wvN  = wv >> 2;           // 2 wave-cols of 64

  const int  rowB  = blockIdx.y * BM;
  const int  colB  = blockIdx.x * BN;
  const long batch = blockIdx.z;

  const float* Ab = A  + batch * bA;
  const float* Bb = Bm + batch * bB;
  float*       Db = D  + batch * bD;

  const v8f zero = {0.f,0.f,0.f,0.f,0.f,0.f,0.f,0.f};
  v8f acc[2][4] = {{zero, zero, zero, zero}, {zero, zero, zero, zero}};

  // ---- register staging for the software pipeline ----
  F4 rA[4];            // A: 4 x float4 per thread  (128x32 tile)
  F4 rB[2][2];         // B: 2 iters x 2 K-adjacent float4 (32x128 tile)

  auto load_stage = [&](int k0) {
    #pragma unroll
    for (int it = 0; it < 4; ++it) {
      int idx = tid + it * TB;         // 1024 quads
      int r = idx >> 3, kq = idx & 7;  // 8 quads per 32-wide row
      int gr = rowB + r, gk = k0 + kq * 4;
      if (gr < M && gk + 3 < K)
        rA[it] = *reinterpret_cast<const F4*>(&Ab[(long)gr * lda + gk]);
      else
        rA[it] = F4{{0.f, 0.f, 0.f, 0.f}};
    }
    #pragma unroll
    for (int it = 0; it < 2; ++it) {   // 512 row-pair quads for 32x128 B tile
      int idx = tid + it * TB;
      int rp = idx >> 5, nq = idx & 31;        // k rows {2rp,2rp+1}, cols nq*4..+3
      int gk = k0 + rp * 2, gn = colB + nq * 4;
      if (gk + 1 < K && gn + 3 < N) {
        rB[it][0] = *reinterpret_cast<const F4*>(&Bb[(long)gk * ldb + gn]);
        rB[it][1] = *reinterpret_cast<const F4*>(&Bb[(long)(gk + 1) * ldb + gn]);
      } else {
        F4 t0{{0.f,0.f,0.f,0.f}}, t1{{0.f,0.f,0.f,0.f}};
        #pragma unroll
        for (int j = 0; j < 4; ++j) {
          if (gk < K && gn + j < N)     t0.v[j] = Bb[(long)gk * ldb + gn + j];
          if (gk + 1 < K && gn + j < N) t1.v[j] = Bb[(long)(gk + 1) * ldb + gn + j];
        }
        rB[it][0] = t0;
        rB[it][1] = t1;
      }
    }
    // light prefetch for the stage after this one (global_prefetch_b8)
    int pk = k0 + BK;
    if (pk + 3 < K) {
      int r = tid >> 1, gr = rowB + r;
      if (gr < M) __builtin_prefetch(&Ab[(long)gr * lda + pk + ((tid & 1) << 4)], 0, 3);
    }
  };

  auto store_stage = [&]() {
    #pragma unroll
    for (int it = 0; it < 4; ++it) {
      int idx = tid + it * TB;
      int r = idx >> 3, kq = idx & 7;
      U2 p;
      p.x = pack2bf(rA[it].v[0], rA[it].v[1]);   // v_cvt_pk_bf16_f32
      p.y = pack2bf(rA[it].v[2], rA[it].v[3]);
      *reinterpret_cast<U2*>(&As[r][kq * 4]) = p;          // ds_store_b64
    }
    #pragma unroll
    for (int it = 0; it < 2; ++it) {
      int idx = tid + it * TB;
      int rp = idx >> 5, nq = idx & 31;
      #pragma unroll
      for (int j = 0; j < 4; ++j) {                        // transpose: pack k,k+1
        unsigned pk = pack2bf(rB[it][0].v[j], rB[it][1].v[j]);
        *reinterpret_cast<unsigned int*>(&BsT[nq * 4 + j][rp * 2]) = pk; // ds_store_b32
      }
    }
  };

  load_stage(0);
  store_stage();
  __syncthreads();

  for (int k0 = 0; k0 < K; k0 += BK) {
    const bool more = (k0 + BK) < K;
    if (more) load_stage(k0 + BK);     // overlap HBM with WMMA below

    // ---- fragment gathers: contiguous 16B chunks -> ds_load_b128 ----
    Frag16 af[2], bfr[4];
    #pragma unroll
    for (int i = 0; i < 2; ++i) {
      int ra = wvM * 32 + i * 16 + lm;                 // A row (M)
      af[i].q[0] = *reinterpret_cast<const U4*>(&As[ra][half * 8]);       // K=half*8..+7
      af[i].q[1] = *reinterpret_cast<const U4*>(&As[ra][16 + half * 8]);  // K=16+half*8..+7
    }
    #pragma unroll
    for (int j = 0; j < 4; ++j) {
      int cc = wvN * 64 + j * 16 + lm;                 // B col (N)
      bfr[j].q[0] = *reinterpret_cast<const U4*>(&BsT[cc][half * 8]);      // K=half*8..+7
      bfr[j].q[1] = *reinterpret_cast<const U4*>(&BsT[cc][16 + half * 8]); // K=16+half*8..+7
    }
    #pragma unroll
    for (int i = 0; i < 2; ++i)
      #pragma unroll
      for (int j = 0; j < 4; ++j)
        acc[i][j] = __builtin_amdgcn_wmma_f32_16x16x32_bf16(
            false, af[i].v, false, bfr[j].v,
            (short)0, acc[i][j], false, false);

    if (more) {
      __syncthreads();                 // all LDS reads of this stage done
      store_stage();
      __syncthreads();                 // next stage visible
    }
  }

  // ---- epilogue: D layout row = v + 8*half, col = lane%16 ----
  #pragma unroll
  for (int i = 0; i < 2; ++i) {
    #pragma unroll
    for (int j = 0; j < 4; ++j) {
      int col = colB + wvN * 64 + j * 16 + lm;
      float bv = (bias && col < N) ? bias[batch * biasB + col] : 0.f;
      #pragma unroll
      for (int v = 0; v < 8; ++v) {
        int row = rowB + wvM * 32 + i * 16 + v + 8 * half;
        if (row < M && col < N) {
          long off = (long)row * ldd + col;
          float val = alpha * acc[i][j][v] + bv;
          if (beta != 0.f) val += beta * Db[off];
          if (act)         val = fmaxf(val, 0.f);
          Db[off] = val;
        }
      }
    }
  }
}

// ---------------------------------------------------------------------------
// Elementwise helpers
// ---------------------------------------------------------------------------
__global__ void k_tables(float* __restrict__ Ct, float* __restrict__ St, int N) {
  long i = (long)blockIdx.x * blockDim.x + threadIdx.x;
  if (i >= (long)N * N) return;
  int k = (int)(i / N), n = (int)(i % N);
  long ph = ((long)k * (long)n) % N;                // exact phase reduction
  float ang = 6.28318530717958647692f * (float)ph / (float)N;
  Ct[i] = cosf(ang);
  St[i] = sinf(ang);
}

__global__ void k_wprep(const float* __restrict__ w, float* __restrict__ Wp,
                        float* __restrict__ Wm, int n) {
  int i = blockIdx.x * blockDim.x + threadIdx.x;
  if (i >= n) return;
  float a = w[i], b = w[n + i];
  Wp[i] = 0.5f * (a + b);
  Wm[i] = 0.5f * (a - b);
}

// radix-2 FFT along the batch axis (size 2), real & imag planes
__global__ void k_butterfly(float* __restrict__ R, float* __restrict__ I, long halfN) {
  long i = (long)blockIdx.x * blockDim.x + threadIdx.x;
  if (i >= halfN) return;
  float r0 = R[i], r1 = R[i + halfN];
  R[i] = r0 + r1; R[i + halfN] = r0 - r1;
  float i0 = I[i], i1 = I[i + halfN];
  I[i] = i0 + i1; I[i + halfN] = i0 - i1;
}

// a[m,c] = Re+Im of spectrum at kept region (all h, w<KEPT)
__global__ void k_gather_a(const float* __restrict__ Ur, const float* __restrict__ Ui,
                           float* __restrict__ a) {
  long idx = (long)blockIdx.x * blockDim.x + threadIdx.x;
  if (idx >= (long)MR * DC) return;
  int c = (int)(idx % DC);
  int m = (int)(idx / DC);
  int w = m % KEPT, t = m / KEPT;
  int h = t % DH, b = t / DH;
  long src = (((long)(b * DH + h)) * DW + w) * DC + c;
  a[idx] = Ur[src] + Ui[src];
}

// an[m,c] = x[b, (H-h)%H, (W-w)%W, c]  (flip + roll)
__global__ void k_gather_an(const float* __restrict__ x, float* __restrict__ an) {
  long idx = (long)blockIdx.x * blockDim.x + threadIdx.x;
  if (idx >= (long)MR * DC) return;
  int c = (int)(idx % DC);
  int m = (int)(idx / DC);
  int w = m % KEPT, t = m / KEPT;
  int h = t % DH, b = t / DH;
  int hh = (DH - h) & (DH - 1);
  int ww = (DW - w) & (DW - 1);
  an[idx] = x[(((long)(b * DH + hh)) * DW + ww) * DC + c];
}

// full spectrum: soft-threshold(o2k+o2nk) in kept region, zero elsewhere
__global__ void k_scatter(const float* __restrict__ o2k, const float* __restrict__ o2nk,
                          float* __restrict__ full) {
  long idx = (long)blockIdx.x * blockDim.x + threadIdx.x;
  if (idx >= NTOT) return;
  int c  = (int)(idx % DC);
  long t = idx / DC;
  int w  = (int)(t % DW);
  long t2 = t / DW;
  int h = (int)(t2 % DH), b = (int)(t2 / DH);
  float v = 0.f;
  if (w < KEPT) {
    long o = ((long)((b * DH + h) * KEPT + w)) * DC + c;
    float s  = o2k[o] + o2nk[o];
    float av = fabsf(s) - 0.01f;                    // SPARSITY
    v = (av > 0.f) ? copysignf(av, s) : 0.f;
  }
  full[idx] = v;
}

__global__ void k_final(const float* __restrict__ Ur, const float* __restrict__ Ui,
                        const float* __restrict__ x, float* __restrict__ out, float inv) {
  long idx = (long)blockIdx.x * blockDim.x + threadIdx.x;
  if (idx >= NTOT) return;
  out[idx] = (Ur[idx] + Ui[idx]) * inv + x[idx];
}

// ---------------------------------------------------------------------------
static inline void gemm(hipStream_t st, const float* A, const float* B, float* D,
                        int M, int N, int K, long lda, long ldb, long ldd,
                        long bA, long bB, long bD, int batches,
                        const float* bias, long biasB,
                        float alpha, float beta, int act) {
  dim3 g((unsigned)((N + BN - 1) / BN), (unsigned)((M + BM - 1) / BM), (unsigned)batches);
  k_gemm<<<g, TB, 0, st>>>(A, B, D, M, N, K, lda, ldb, ldd, bA, bB, bD,
                           bias, biasB, alpha, beta, act);
}

static inline dim3 eg(long n) { return dim3((unsigned)((n + 255) / 256)); }

extern "C" void kernel_launch(void* const* d_in, const int* in_sizes, int n_in,
                              void* d_out, int out_size, void* d_ws, size_t ws_size,
                              hipStream_t stream) {
  (void)in_sizes; (void)n_in; (void)out_size; (void)ws_size;
  const float* x  = (const float*)d_in[0];
  const float* w1 = (const float*)d_in[1];   // (2, 8, 96, 96)
  const float* b1 = (const float*)d_in[2];   // (2, 8, 96)
  const float* w2 = (const float*)d_in[3];
  const float* b2 = (const float*)d_in[4];
  float* out = (float*)d_out;

  // ---- workspace carve ----
  float* p = (float*)d_ws;
  float* Ur = p; p += NTOT;
  float* Ui = p; p += NTOT;
  float* Vr = p; p += NTOT;
  float* Vi = p; p += NTOT;
  float* T768c = p; p += 768 * 768;
  float* T768s = p; p += 768 * 768;
  float* T256c = p; p += 256 * 256;
  float* T256s = p; p += 256 * 256;
  float* T128c = p; p += 128 * 128;
  float* T128s = p; p += 128 * 128;
  const int WSZ = 8 * 96 * 96;
  float* W1p = p; p += WSZ;
  float* W1m = p; p += WSZ;
  float* W2p = p; p += WSZ;
  float* W2m = p; p += WSZ;
  const long MRC = (long)MR * DC;
  float* a_b  = p; p += MRC;
  float* an_b = p; p += MRC;
  float* o1k  = p; p += MRC;
  float* o1nk = p; p += MRC;
  float* o2k  = p; p += MRC;
  float* o2nk = p; p += MRC;

  // ---- DFT tables (symmetric matrices) + Hartley-combined weights ----
  k_tables<<<eg(768L*768), 256, 0, stream>>>(T768c, T768s, 768);
  k_tables<<<eg(256L*256), 256, 0, stream>>>(T256c, T256s, 256);
  k_tables<<<eg(128L*128), 256, 0, stream>>>(T128c, T128s, 128);
  k_wprep<<<eg(WSZ), 256, 0, stream>>>(w1, W1p, W1m, WSZ);
  k_wprep<<<eg(WSZ), 256, 0, stream>>>(w2, W2p, W2m, WSZ);

  const int  MBH = DB * DH * DW;          // 65536 rows for axis-C GEMMs
  const long sW  = (long)DW * DC;         // W-axis slab stride
  const int  NH  = DW * DC;               // 196608 cols for H-axis GEMMs
  const long sH  = (long)DH * NH;

  // ================= forward DHT =================
  // axis C (real input): yr = x*C, yi = -x*S
  gemm(stream, x, T768c, Ur, MBH, DC, DC, DC, DC, DC, 0,0,0, 1, nullptr,0,  1.f, 0.f, 0);
  gemm(stream, x, T768s, Ui, MBH, DC, DC, DC, DC, DC, 0,0,0, 1, nullptr,0, -1.f, 0.f, 0);
  // axis W (left-multiply, batched over B*H slabs): Vr=C*Ur+S*Ui ; Vi=C*Ui-S*Ur
  gemm(stream, T256c, Ur, Vr, DW, DC, DW, DW, DC, DC, 0, sW, sW, DB*DH, nullptr,0,  1.f, 0.f, 0);
  gemm(stream, T256s, Ui, Vr, DW, DC, DW, DW, DC, DC, 0, sW, sW, DB*DH, nullptr,0,  1.f, 1.f, 0);
  gemm(stream, T256c, Ui, Vi, DW, DC, DW, DW, DC, DC, 0, sW, sW, DB*DH, nullptr,0,  1.f, 0.f, 0);
  gemm(stream, T256s, Ur, Vi, DW, DC, DW, DW, DC, DC, 0, sW, sW, DB*DH, nullptr,0, -1.f, 1.f, 0);
  // axis H (batched over B): Ur=C*Vr+S*Vi ; Ui=C*Vi-S*Vr
  gemm(stream, T128c, Vr, Ur, DH, NH, DH, DH, NH, NH, 0, sH, sH, DB, nullptr,0,  1.f, 0.f, 0);
  gemm(stream, T128s, Vi, Ur, DH, NH, DH, DH, NH, NH, 0, sH, sH, DB, nullptr,0,  1.f, 1.f, 0);
  gemm(stream, T128c, Vi, Ui, DH, NH, DH, DH, NH, NH, 0, sH, sH, DB, nullptr,0,  1.f, 0.f, 0);
  gemm(stream, T128s, Vr, Ui, DH, NH, DH, DH, NH, NH, 0, sH, sH, DB, nullptr,0, -1.f, 1.f, 0);
  // axis B (size 2): radix-2 butterfly in place
  k_butterfly<<<eg(NTOT/2), 256, 0, stream>>>(Ur, Ui, NTOT / 2);

  // kept region gather (+Hartley Re+Im combine) and spatial flip term
  k_gather_a <<<eg(MRC), 256, 0, stream>>>(Ur, Ui, a_b);
  k_gather_an<<<eg(MRC), 256, 0, stream>>>(x, an_b);

  // ============ block-diagonal MLP (batched over 8 blocks) ============
  // o1_k = relu(a*W1p + an*W1m + b1[0]) ; o1_nk = relu(an*W1p + a*W1m + b1[1])
  gemm(stream, a_b,  W1p, o1k,  MR, 96, 96, DC, 96, DC, 96, 96*96, 96, 8, nullptr, 0, 1.f, 0.f, 0);
  gemm(stream, an_b, W1m, o1k,  MR, 96, 96, DC, 96, DC, 96, 96*96, 96, 8, b1,     96, 1.f, 1.f, 1);
  gemm(stream, an_b, W1p, o1nk, MR, 96, 96, DC, 96, DC, 96, 96*96, 96, 8, nullptr, 0, 1.f, 0.f, 0);
  gemm(stream, a_b,  W1m, o1nk, MR, 96, 96, DC, 96, DC, 96, 96*96, 96, 8, b1+768, 96, 1.f, 1.f, 1);
  // o2_k = o1k*W2p + o1nk*W2m + b2[0] ; o2_nk = o1nk*W2p + o2k*W2m + b2[1]
  gemm(stream, o1k,  W2p, o2k,  MR, 96, 96, DC, 96, DC, 96, 96*96, 96, 8, nullptr, 0, 1.f, 0.f, 0);
  gemm(stream, o1nk, W2m, o2k,  MR, 96, 96, DC, 96, DC, 96, 96*96, 96, 8, b2,     96, 1.f, 1.f, 0);
  gemm(stream, o1nk, W2p, o2nk, MR, 96, 96, DC, 96, DC, 96, 96*96, 96, 8, nullptr, 0, 1.f, 0.f, 0);
  gemm(stream, o2k,  W2m, o2nk, MR, 96, 96, DC, 96, DC, 96, 96*96, 96, 8, b2+768, 96, 1.f, 1.f, 0);

  // soft-threshold + zero-padded scatter (full spectrum lives in Vr)
  k_scatter<<<eg(NTOT), 256, 0, stream>>>(o2k, o2nk, Vr);

  // ================= inverse DHT (forward DHT / size) =================
  gemm(stream, Vr, T768c, Ur, MBH, DC, DC, DC, DC, DC, 0,0,0, 1, nullptr,0,  1.f, 0.f, 0);
  gemm(stream, Vr, T768s, Ui, MBH, DC, DC, DC, DC, DC, 0,0,0, 1, nullptr,0, -1.f, 0.f, 0);
  gemm(stream, T256c, Ur, Vr, DW, DC, DW, DW, DC, DC, 0, sW, sW, DB*DH, nullptr,0,  1.f, 0.f, 0);
  gemm(stream, T256s, Ui, Vr, DW, DC, DW, DW, DC, DC, 0, sW, sW, DB*DH, nullptr,0,  1.f, 1.f, 0);
  gemm(stream, T256c, Ui, Vi, DW, DC, DW, DW, DC, DC, 0, sW, sW, DB*DH, nullptr,0,  1.f, 0.f, 0);
  gemm(stream, T256s, Ur, Vi, DW, DC, DW, DW, DC, DC, 0, sW, sW, DB*DH, nullptr,0, -1.f, 1.f, 0);
  gemm(stream, T128c, Vr, Ur, DH, NH, DH, DH, NH, NH, 0, sH, sH, DB, nullptr,0,  1.f, 0.f, 0);
  gemm(stream, T128s, Vi, Ur, DH, NH, DH, DH, NH, NH, 0, sH, sH, DB, nullptr,0,  1.f, 1.f, 0);
  gemm(stream, T128c, Vi, Ui, DH, NH, DH, DH, NH, NH, 0, sH, sH, DB, nullptr,0,  1.f, 0.f, 0);
  gemm(stream, T128s, Vr, Ui, DH, NH, DH, DH, NH, NH, 0, sH, sH, DB, nullptr,0, -1.f, 1.f, 0);
  k_butterfly<<<eg(NTOT/2), 256, 0, stream>>>(Ur, Ui, NTOT / 2);

  // out = (Re+Im)/N + x
  k_final<<<eg(NTOT), 256, 0, stream>>>(Ur, Ui, x, out, 1.f / (float)NTOT);
}
